// Coref_Injection_52682068853221
// MI455X (gfx1250) — compile-verified, matrix operand order checked by reading, not run on hardware
//
#include <hip/hip_runtime.h>
#include <hip/hip_bf16.h>

// Problem dims (fixed by the reference)
constexpr int Bn = 4, Tn = 2048, Dn = 256, Mn = 128, Ln = 16, Pn = Mn * Ln; // P=2048

typedef __attribute__((ext_vector_type(16))) __bf16        bf16x16;
typedef __attribute__((ext_vector_type(8)))  float         f32x8;
typedef __attribute__((ext_vector_type(4)))  unsigned int  u32x4;
typedef __attribute__((ext_vector_type(8)))  int           i32x8;
typedef __attribute__((ext_vector_type(4)))  int           i32x4;

// Padded LDS row stride for TDM-staged f32 tiles (pad 1 DWORD / 256 DWORDs)
constexpr int XS = 257;

// D# group1 word0 flags (ISA 8.4): data_size=2 (4B) at [17:16]; pad_enable
// bit20; pad_interval [24:22] (7 => 256 DWORDs); pad_amount [31:25] (0 => 1).
constexpr unsigned TDM_F32     = (2u << 16);
constexpr unsigned TDM_F32_PAD = (2u << 16) | (1u << 20) | (7u << 22);

__device__ inline unsigned lds_addr_of(const void* p) {
  // Generic LDS pointers carry the LDS byte offset in the low 32 bits.
  return (unsigned)(unsigned long long)(uintptr_t)p;
}

// Issue a 2D TDM tile load Global->LDS (TENSOR_LOAD_TO_LDS, ISA ch.7/8).
// Declared tensor dims == tile dims -> no OOB clamping. Groups 2/3 zero (2D).
__device__ inline void tdm_load_2d(unsigned lds_byte_addr, const void* gptr,
                                   unsigned tile_d0, unsigned tile_d1,
                                   unsigned long long stride_d0_elems,
                                   unsigned flags_word0) {
  unsigned long long ga = (unsigned long long)(uintptr_t)gptr;
  u32x4 g0;
  g0[0] = 1u;                                                // count=1, user D#
  g0[1] = lds_byte_addr;                                     // lds_addr
  g0[2] = (unsigned)(ga & 0xffffffffu);                      // global_addr[31:0]
  g0[3] = (unsigned)((ga >> 32) & 0x01ffffffu) | (2u << 30); // addr[56:32]|type=2
  i32x8 g1;
  g1[0] = (int)flags_word0;
  g1[1] = (int)((tile_d0 & 0xffffu) << 16);                  // tensor_dim0[15:0]
  g1[2] = (int)(((tile_d0 >> 16) & 0xffffu) |
                ((tile_d1 & 0xffffu) << 16));                // td0[31:16]|td1[15:0]
  g1[3] = (int)(((tile_d1 >> 16) & 0xffffu) |
                ((tile_d0 & 0xffffu) << 16));                // td1[31:16]|tile_dim0
  g1[4] = (int)(tile_d1 & 0xffffu);                          // tile_dim1; tile_dim2=0
  g1[5] = (int)(stride_d0_elems & 0xffffffffu);
  g1[6] = (int)((stride_d0_elems >> 32) & 0xffffu);
  g1[7] = 0;
  i32x4 z4 = {0, 0, 0, 0};
#if defined(__clang_major__) && __clang_major__ >= 23
  i32x8 z8 = {0, 0, 0, 0, 0, 0, 0, 0};
  __builtin_amdgcn_tensor_load_to_lds(g0, g1, z4, z4, z8, 0);
#else
  __builtin_amdgcn_tensor_load_to_lds(g0, g1, z4, z4, 0);
#endif
}

// --- WMMA fragment builders (ISA 7.12.2 layout) ----------------------------
// Lane l<16 -> row/col l; K = {kb..kb+7, 16+kb..23+kb}, kb = 8*(lane>=16).

// A/B fragment from a row-major bf16 LDS tile [16][32].
__device__ inline bf16x16 load_frag_lds(const __bf16* base, int lane) {
  int row = lane & 15;
  int kb  = (lane >> 4) << 3;
  const __bf16* p = base + row * 32 + kb;
  bf16x16 f;
#pragma unroll
  for (int i = 0; i < 8; ++i) f[i] = p[i];
#pragma unroll
  for (int i = 0; i < 8; ++i) f[8 + i] = p[16 + i];
  return f;
}

// A fragment from a row-major f32 LDS tile [16][32] (TDM-staged), cvt->bf16.
__device__ inline bf16x16 frag_a_f32(const float* base, int lane) {
  int row = lane & 15;
  int kb  = (lane >> 4) << 3;
  const float* p = base + row * 32 + kb;
  bf16x16 f;
#pragma unroll
  for (int i = 0; i < 8; ++i) f[i] = (__bf16)p[i];
#pragma unroll
  for (int i = 0; i < 8; ++i) f[8 + i] = (__bf16)p[16 + i];
  return f;
}

// B fragment from a row-major f32 LDS tile [32][XS] (TDM-staged, padded):
// lane picks column col0 + (lane&15), reads its 16 K values down the rows.
__device__ inline bf16x16 frag_b_f32p(const float* tile, int col0, int lane) {
  int c  = lane & 15;
  int kb = (lane >> 4) << 3;
  const float* p = tile + col0 + c;
  bf16x16 f;
#pragma unroll
  for (int i = 0; i < 8; ++i) f[i] = (__bf16)p[(kb + i) * XS];
#pragma unroll
  for (int i = 0; i < 8; ++i) f[8 + i] = (__bf16)p[(16 + kb + i) * XS];
  return f;
}

// ---------------------------------------------------------------------------
// Kernel 1: head_rep AND tail_rep = [P x T] @ [T x D], fused.
// One block produces 16 head rows + 16 tail rows sharing the same x tiles:
// halves x HBM traffic and halves B-fragment LDS reads per FLOP.
// Double-buffered TDM (3 descriptors / K-step), bf16 WMMA, f32 accumulate.
// grid = B * P/16 = 512, block = 128 (4 waves, 64 cols each)
// ---------------------------------------------------------------------------
__global__ __launch_bounds__(128)
void rep_gemm_kernel(const float* __restrict__ head, const float* __restrict__ tail,
                     const float* __restrict__ x,
                     float* __restrict__ head_rep, float* __restrict__ tail_rep) {
  __shared__ float AfH[2][16 * 32];  // head tile (f32)          4 KB
  __shared__ float AfT[2][16 * 32];  // tail tile (f32)          4 KB
  __shared__ float Xf[2][32 * XS];   // x tile 32K x 256N padded ~66 KB

  const int tid   = threadIdx.x;
  const int lane  = tid & 31;
  const int wave  = tid >> 5;
  const int nbase = wave * 64;

  const int b  = blockIdx.x >> 7;        // P/16 == 128
  const int rt = blockIdx.x & 127;

  const float* hrow = head + ((size_t)b * Pn + (size_t)rt * 16) * Tn;
  const float* trow = tail + ((size_t)b * Pn + (size_t)rt * 16) * Tn;
  const float* xb   = x    + (size_t)b * Tn * Dn;

  f32x8 accH[4] = {};
  f32x8 accT[4] = {};

  constexpr int NK = Tn / 32;            // 64 K-steps
  if (tid == 0) {                        // prologue: tile 0 in flight
    tdm_load_2d(lds_addr_of(AfH[0]), hrow, 32, 16, (unsigned long long)Tn, TDM_F32);
    tdm_load_2d(lds_addr_of(AfT[0]), trow, 32, 16, (unsigned long long)Tn, TDM_F32);
    tdm_load_2d(lds_addr_of(Xf[0]),  xb,  256, 32, (unsigned long long)Dn, TDM_F32_PAD);
  }

  for (int i = 0; i < NK; ++i) {
    const int cur = i & 1;
    if (tid == 0) {
      if (i + 1 < NK) {
        const int k1 = (i + 1) * 32;
        tdm_load_2d(lds_addr_of(AfH[cur ^ 1]), hrow + k1, 32, 16,
                    (unsigned long long)Tn, TDM_F32);
        tdm_load_2d(lds_addr_of(AfT[cur ^ 1]), trow + k1, 32, 16,
                    (unsigned long long)Tn, TDM_F32);
        tdm_load_2d(lds_addr_of(Xf[cur ^ 1]), xb + (size_t)k1 * Dn, 256, 32,
                    (unsigned long long)Dn, TDM_F32_PAD);
        __builtin_amdgcn_s_wait_tensorcnt(3);  // current tile's 3 loads done
      } else {
        __builtin_amdgcn_s_wait_tensorcnt(0);
      }
    }
    __syncthreads();                     // publish TDM-written LDS to all waves

    // Build ALL fragments first, then issue the WMMA batch back-to-back:
    // pays the WMMA<->VALU WAR hazard once per K-step instead of per-WMMA.
    bf16x16 ah = frag_a_f32(AfH[cur], lane);
    bf16x16 at = frag_a_f32(AfT[cur], lane);
    bf16x16 bf[4];
#pragma unroll
    for (int t = 0; t < 4; ++t) bf[t] = frag_b_f32p(Xf[cur], nbase + t * 16, lane);
#pragma unroll
    for (int t = 0; t < 4; ++t)
      accH[t] = __builtin_amdgcn_wmma_f32_16x16x32_bf16(
          false, ah, false, bf[t], (short)0, accH[t], false, false);
#pragma unroll
    for (int t = 0; t < 4; ++t)
      accT[t] = __builtin_amdgcn_wmma_f32_16x16x32_bf16(
          false, at, false, bf[t], (short)0, accT[t], false, false);
    __syncthreads();                     // retire buffer before TDM reuses it
  }

  float* dH = head_rep + ((size_t)b * Pn + (size_t)rt * 16) * Dn;
  float* dT = tail_rep + ((size_t)b * Pn + (size_t)rt * 16) * Dn;
#pragma unroll
  for (int t = 0; t < 4; ++t) {
#pragma unroll
    for (int r = 0; r < 8; ++r) {
      int row = r + ((lane >> 4) << 3);
      int col = nbase + t * 16 + (lane & 15);
      dH[(size_t)row * Dn + col] = accH[t][r];
      dT[(size_t)row * Dn + col] = accT[t][r];
    }
  }
}

// ---------------------------------------------------------------------------
// Kernel 2: logits = relu([hr,tr,hr*tr] @ w1 + b1) @ w2 + b2
// A (feat) staged manually (computed on the fly); w1 tile via TDM.
// grid = B*P/16 = 512, block = 128
// ---------------------------------------------------------------------------
__global__ __launch_bounds__(128)
void mlp_kernel(const float* __restrict__ head_rep, const float* __restrict__ tail_rep,
                const float* __restrict__ w1, const float* __restrict__ b1,
                const float* __restrict__ w2, const float* __restrict__ b2,
                float* __restrict__ logits) {
  __shared__ __bf16 As[16 * 32];     // feat tile (bf16)
  __shared__ float  Wf[32 * XS];     // w1 tile via TDM (f32, padded)
  __shared__ float  Hs[16 * 256];    // relu hidden

  const int tid   = threadIdx.x;
  const int lane  = tid & 31;
  const int wave  = tid >> 5;
  const int nbase = wave * 64;
  const int p0    = blockIdx.x * 16;   // row tile in flattened [B*P]

  f32x8 acc[4] = {};

  for (int k0 = 0; k0 < 3 * Dn; k0 += 32) {
    // Stage feat tile on the fly: feat[p, k] from head/tail reps
    for (int i = tid; i < 16 * 32; i += 128) {
      int r = i >> 5, k = i & 31;
      int kk = k0 + k;
      size_t prow = (size_t)(p0 + r) * Dn;
      float v;
      if (kk < Dn)          v = head_rep[prow + kk];
      else if (kk < 2 * Dn) v = tail_rep[prow + kk - Dn];
      else                  v = head_rep[prow + kk - 2 * Dn] * tail_rep[prow + kk - 2 * Dn];
      As[i] = (__bf16)v;
    }
    // Stage w1 K-rows [32][256] via TDM (contiguous rows, stride Dn)
    if (tid == 0) {
      tdm_load_2d(lds_addr_of(Wf), w1 + (size_t)k0 * Dn, 256, 32,
                  (unsigned long long)Dn, TDM_F32_PAD);
      __builtin_amdgcn_s_wait_tensorcnt(0);
    }
    __syncthreads();

    bf16x16 a = load_frag_lds(As, lane);
    bf16x16 wf[4];
#pragma unroll
    for (int t = 0; t < 4; ++t) wf[t] = frag_b_f32p(Wf, nbase + t * 16, lane);
#pragma unroll
    for (int t = 0; t < 4; ++t)
      acc[t] = __builtin_amdgcn_wmma_f32_16x16x32_bf16(
          false, a, false, wf[t], (short)0, acc[t], false, false);
    __syncthreads();
  }

  // h = relu(acc + b1) into LDS (f32)
#pragma unroll
  for (int t = 0; t < 4; ++t) {
#pragma unroll
    for (int r = 0; r < 8; ++r) {
      int row = r + ((lane >> 4) << 3);
      int col = nbase + t * 16 + (lane & 15);
      float hv = acc[t][r] + b1[col];
      Hs[row * 256 + col] = hv > 0.0f ? hv : 0.0f;
    }
  }
  __syncthreads();

  // logits[p0+row, j] = Hs[row,:] . w2[:,j] + b2[j]   (32 (row,j) pairs)
  if (tid < 32) {
    int row = tid >> 1, j = tid & 1;
    float s = b2[j];
#pragma unroll 8
    for (int d = 0; d < Dn; ++d) s += Hs[row * 256 + d] * w2[d * 2 + j];
    logits[(size_t)(p0 + row) * 2 + j] = s;
  }
}

// ---------------------------------------------------------------------------
// Kernel 3: per-mention top-1 over L=16, scale selected tail_rep row
// grid = B*M = 512, block = 256 (== D)
// ---------------------------------------------------------------------------
__global__ __launch_bounds__(256)
void select_kernel(const float* __restrict__ logits, const float* __restrict__ tail_rep,
                   float* __restrict__ rep_m) {
  __shared__ float sc[Ln];
  __shared__ float bval;
  __shared__ int   bidx;

  const int bm  = blockIdx.x;        // b*M + m ; pair base = bm*L (P = M*L)
  const int tid = threadIdx.x;

  if (tid < Ln) sc[tid] = logits[((size_t)bm * Ln + tid) * 2 + 1];
  __syncthreads();
  if (tid == 0) {
    float bv = sc[0]; int bi = 0;
    for (int l = 1; l < Ln; ++l)
      if (sc[l] > bv) { bv = sc[l]; bi = l; }   // strict > -> first-max
    bval = bv; bidx = bi;
  }
  __syncthreads();

  rep_m[(size_t)bm * Dn + tid] =
      tail_rep[((size_t)bm * Ln + bidx) * Dn + tid] * bval;
}

// ---------------------------------------------------------------------------
// Kernel 4: out = x + [T x M] @ [M x D]  (A[t,m] = pos[b,m,t]; rep_m via TDM)
// grid = B * T/16 = 512, block = 128
// ---------------------------------------------------------------------------
__global__ __launch_bounds__(128)
void scatter_gemm_kernel(const float* __restrict__ pos, const float* __restrict__ rep_m,
                         const float* __restrict__ x, float* __restrict__ out) {
  __shared__ __bf16 As[16 * 32];     // pos tile transposed (bf16)
  __shared__ float  Bf[32 * XS];     // rep_m tile via TDM (f32, padded)

  const int tid   = threadIdx.x;
  const int lane  = tid & 31;
  const int wave  = tid >> 5;
  const int nbase = wave * 64;

  const int b  = blockIdx.x >> 7;       // T/16 == 128
  const int t0 = (blockIdx.x & 127) * 16;

  f32x8 acc[4] = {};

  for (int m0 = 0; m0 < Mn; m0 += 32) {
    // A[r][k] = pos[b, m0+k, t0+r] (consecutive lanes -> consecutive t)
    for (int i = tid; i < 16 * 32; i += 128) {
      int k = i >> 4, r = i & 15;
      As[r * 32 + k] = (__bf16)pos[((size_t)b * Mn + m0 + k) * Tn + t0 + r];
    }
    // rep_m K-rows [32][256] are fully contiguous: TDM load
    if (tid == 0) {
      tdm_load_2d(lds_addr_of(Bf), rep_m + ((size_t)b * Mn + m0) * Dn, 256, 32,
                  (unsigned long long)Dn, TDM_F32_PAD);
      __builtin_amdgcn_s_wait_tensorcnt(0);
    }
    __syncthreads();

    bf16x16 a = load_frag_lds(As, lane);
    bf16x16 bfr[4];
#pragma unroll
    for (int t = 0; t < 4; ++t) bfr[t] = frag_b_f32p(Bf, nbase + t * 16, lane);
#pragma unroll
    for (int t = 0; t < 4; ++t)
      acc[t] = __builtin_amdgcn_wmma_f32_16x16x32_bf16(
          false, a, false, bfr[t], (short)0, acc[t], false, false);
    __syncthreads();
  }

#pragma unroll
  for (int t = 0; t < 4; ++t) {
#pragma unroll
    for (int r = 0; r < 8; ++r) {
      int row = r + ((lane >> 4) << 3);
      int col = nbase + t * 16 + (lane & 15);
      size_t idx = ((size_t)b * Tn + t0 + row) * Dn + col;
      out[idx] = x[idx] + acc[t][r];
    }
  }
}

// ---------------------------------------------------------------------------
// Kernel 5: masked KL loss partial sums -> global atomic f32 accumulators
// grid = B*P/256 = 32, block = 256
// ---------------------------------------------------------------------------
__global__ __launch_bounds__(256)
void loss_kernel(const float* __restrict__ logits, const float* __restrict__ lab,
                 const unsigned char* __restrict__ mask, float* __restrict__ accum) {
  __shared__ float rs[256];
  __shared__ float rc[256];

  const int tid  = threadIdx.x;
  const int flat = blockIdx.x * 256 + tid;   // pair index in [0, B*P)

  float s = 0.0f, c = 0.0f;
  if (mask[flat]) {
    float l0 = logits[(size_t)flat * 2 + 0];
    float l1 = logits[(size_t)flat * 2 + 1];
    float mx = fmaxf(l0, l1);
    float lz = mx + __logf(__expf(l0 - mx) + __expf(l1 - mx));
#pragma unroll
    for (int j = 0; j < 2; ++j) {
      float la = lab[(size_t)flat * 2 + j];
      float lq = (j ? l1 : l0) - lz;
      float self = (la > 0.0f) ? la * __logf(fmaxf(la, 1e-38f)) : 0.0f;
      s += self - la * lq;
    }
    c = 1.0f;
  }
  rs[tid] = s; rc[tid] = c;
  __syncthreads();
  for (int off = 128; off > 0; off >>= 1) {
    if (tid < off) { rs[tid] += rs[tid + off]; rc[tid] += rc[tid + off]; }
    __syncthreads();
  }
  if (tid == 0) {
    atomicAdd(&accum[0], rs[0]);
    atomicAdd(&accum[1], rc[0]);
  }
}

__global__ void init_accum_kernel(float* accum) {
  accum[0] = 0.0f; accum[1] = 0.0f;
}

__global__ void finalize_loss_kernel(const float* __restrict__ accum,
                                     float* __restrict__ out_loss) {
  out_loss[0] = accum[0] / (accum[1] * 2.0f);
}

// ---------------------------------------------------------------------------
// Launcher
// ---------------------------------------------------------------------------
extern "C" void kernel_launch(void* const* d_in, const int* in_sizes, int n_in,
                              void* d_out, int out_size, void* d_ws, size_t ws_size,
                              hipStream_t stream) {
  const float*         head = (const float*)d_in[0];
  const float*         tail = (const float*)d_in[1];
  /* lens (d_in[2]) unused: uniform segment lengths == L */
  const float*         x    = (const float*)d_in[3];
  const float*         pos  = (const float*)d_in[4];
  const float*         lab  = (const float*)d_in[5];
  const unsigned char* mask = (const unsigned char*)d_in[6];
  const float*         w1   = (const float*)d_in[7];
  const float*         b1   = (const float*)d_in[8];
  const float*         w2   = (const float*)d_in[9];
  const float*         b2   = (const float*)d_in[10];

  float* out = (float*)d_out;            // [B*T*D] output, then [1] loss

  // Workspace layout (floats)
  float* ws       = (float*)d_ws;
  float* head_rep = ws;                                   // B*P*D = 2097152
  float* tail_rep = head_rep + (size_t)Bn * Pn * Dn;      // 2097152
  float* logits   = tail_rep + (size_t)Bn * Pn * Dn;      // B*P*2 = 16384
  float* rep_m    = logits   + (size_t)Bn * Pn * 2;       // B*M*D = 131072
  float* accum    = rep_m    + (size_t)Bn * Mn * Dn;      // 2

  init_accum_kernel<<<1, 1, 0, stream>>>(accum);

  rep_gemm_kernel<<<Bn * (Pn / 16), 128, 0, stream>>>(
      head, tail, x, head_rep, tail_rep);

  mlp_kernel<<<(Bn * Pn) / 16, 128, 0, stream>>>(
      head_rep, tail_rep, w1, b1, w2, b2, logits);

  select_kernel<<<Bn * Mn, 256, 0, stream>>>(logits, tail_rep, rep_m);

  scatter_gemm_kernel<<<Bn * (Tn / 16), 128, 0, stream>>>(pos, rep_m, x, out);

  loss_kernel<<<(Bn * Pn) / 256, 256, 0, stream>>>(logits, lab, mask, accum);

  finalize_loss_kernel<<<1, 1, 0, stream>>>(accum, out + (size_t)Bn * Tn * Dn);
}
